// Encoder_71528385347709
// MI455X (gfx1250) — matrix operand order repaired and checked by reading, not compiled, not run on hardware
//
#include <hip/hip_runtime.h>
#include <hip/hip_bf16.h>

#define GN 50000
#define GE 400000
#define GB 64
#define GC 128
#define GDEPTH 3

typedef __attribute__((ext_vector_type(16))) __bf16 bf16x16;
typedef __attribute__((ext_vector_type(8)))  float  v8f;

union FragU { unsigned u[8]; bf16x16 v; };

__device__ __forceinline__ unsigned short f2bf(float f) {
    unsigned u = __float_as_uint(f);
    unsigned r = u + 0x7FFFu + ((u >> 16) & 1u);   // round-to-nearest-even
    return (unsigned short)(r >> 16);
}
// order-preserving float->uint encoding for atomicMax-based segment max
__device__ __forceinline__ unsigned enc_ord(float f) {
    unsigned u = __float_as_uint(f);
    return (u & 0x80000000u) ? ~u : (u | 0x80000000u);
}
__device__ __forceinline__ float dec_ord(unsigned e) {
    unsigned u = (e & 0x80000000u) ? (e & 0x7FFFFFFFu) : ~e;
    return __uint_as_float(u);
}

__global__ void zero_f32_kernel(float* p, int n) {
    int t = blockIdx.x * blockDim.x + threadIdx.x;
    if (t < n) p[t] = 0.0f;
}

__global__ void counts_kernel(const int* __restrict__ bid, float* counts) {
    int t = blockIdx.x * blockDim.x + threadIdx.x;
    if (t < GN) atomicAdd(counts + bid[t], 1.0f);
}

// Pack W[l][m][k][n] (f32, row-major KxN, "x @ W") into bf16 B-fragment order:
// ushort index t decomposes as  h:1 | j:3 | lane:5 | kk:2 | ct:3 | m:2 | l:2 (low->high)
// value = W[k][n] with n = ct*16 + (lane&15), k = kk*32 + (lane>>4)*16 + 2j + h
__global__ void pack_weights_kernel(const float* __restrict__ Wq, const float* __restrict__ Wk,
                                    const float* __restrict__ Wv, const float* __restrict__ Ws,
                                    unsigned short* __restrict__ wp) {
    int t = blockIdx.x * blockDim.x + threadIdx.x;
    if (t >= GDEPTH * 4 * GC * GC) return;
    int h    =  t        & 1;
    int j    = (t >> 1)  & 7;
    int lane = (t >> 4)  & 31;
    int kk   = (t >> 9)  & 3;
    int ct   = (t >> 11) & 7;
    int m    = (t >> 14) & 3;
    int l    =  t >> 16;
    int n = ct * 16 + (lane & 15);
    int k = kk * 32 + ((lane >> 4) << 4) + j * 2 + h;
    const float* W = (m == 0) ? Wq : (m == 1) ? Wk : (m == 2) ? Wv : Ws;
    wp[t] = f2bf(W[((size_t)l * GC + k) * GC + n]);
}

__global__ void encoder_kernel(const float* __restrict__ x, const float* __restrict__ eW,
                               const float* __restrict__ eb,
                               float* __restrict__ h, unsigned short* __restrict__ hbf) {
    int t = blockIdx.x * blockDim.x + threadIdx.x;
    if (t >= GN * GC) return;
    int n = t >> 7, c = t & 127;
    float val;
    if (c < 4) {
        val = x[n * 4 + c];
    } else {
        int jc = c - 4;
        val = eb[jc];
#pragma unroll
        for (int i = 0; i < 4; ++i) val += x[n * 4 + i] * eW[i * (GC - 4) + jc];
    }
    h[t] = val;
    hbf[t] = f2bf(val);
}

__global__ void init_layer_kernel(float* denom, unsigned* amax) {
    int t = blockIdx.x * blockDim.x + threadIdx.x;
    if (t < GN) { denom[t] = 0.0f; amax[t] = 0u; }
}

// One block = 32 node rows; wave m in {0..3} computes the 32x128 tile of matrix m
// (q,k,v,skip). Per K-step: 2 A frags + 8 preloaded B frags -> 16 chained WMMAs,
// so each B fragment is amortized over 32 output rows (1.25 loads per v_wmma).
__global__ __launch_bounds__(128) void gemm_qkvs_kernel(
        const unsigned short* __restrict__ hbf, const unsigned short* __restrict__ wp, int layer,
        float* __restrict__ q, float* __restrict__ k, float* __restrict__ v, float* __restrict__ h,
        const float* __restrict__ bq, const float* __restrict__ bk,
        const float* __restrict__ bv, const float* __restrict__ bs) {
    int lane = threadIdx.x & 31;
    int m = threadIdx.x >> 5;
    int row0 = blockIdx.x << 5;                    // 32 rows per block
    float* outp = (m == 0) ? q : (m == 1) ? k : (m == 2) ? v : h;
    const float* bias = (m == 0) ? bq : (m == 1) ? bk : (m == 2) ? bv : bs;

    int lane15 = lane & 15;
    int khalf8 = (lane >> 4) << 3;                 // half of A frag K offset; A uses contiguous k = 16*(lane>>4) + s
    const unsigned short* wbase = wp + (size_t)(layer * 4 + m) * 16384;

    bool has2 = (row0 + 31) < GN;                  // second 16-row tile in range?
    int r0 = row0 + lane15;
    int r1 = has2 ? (r0 + 16) : r0;                // clamp OOB loads to tile 0

    v8f acc0[8] = {};
    v8f acc1[8] = {};
#pragma unroll
    for (int kk = 0; kk < 4; ++kk) {
        FragU a0, a1;
#pragma unroll
        for (int j = 0; j < 8; ++j) {
            int kidx = (kk << 5) + (khalf8 << 1) + (j << 1);
            a0.u[j] = *(const unsigned*)(hbf + (size_t)r0 * GC + kidx);
            a1.u[j] = *(const unsigned*)(hbf + (size_t)r1 * GC + kidx);
        }
        FragU b[8];
#pragma unroll
        for (int ct = 0; ct < 8; ++ct) {
            const uint4* bp = (const uint4*)(wbase + (((size_t)ct * 4 + kk) * 32 + lane) * 16);
            uint4 p0 = bp[0], p1 = bp[1];
            b[ct].u[0] = p0.x; b[ct].u[1] = p0.y; b[ct].u[2] = p0.z; b[ct].u[3] = p0.w;
            b[ct].u[4] = p1.x; b[ct].u[5] = p1.y; b[ct].u[6] = p1.z; b[ct].u[7] = p1.w;
        }
#pragma unroll
        for (int ct = 0; ct < 8; ++ct) {
            acc0[ct] = __builtin_amdgcn_wmma_f32_16x16x32_bf16(
                false, a0.v, false, b[ct].v, (short)0, acc0[ct], false, false);
            acc1[ct] = __builtin_amdgcn_wmma_f32_16x16x32_bf16(
                false, a1.v, false, b[ct].v, (short)0, acc1[ct], false, false);
        }
    }
    int rowadd = (lane >> 4) << 3;
#pragma unroll
    for (int ct = 0; ct < 8; ++ct) {
        int col = (ct << 4) + lane15;
        float bb = bias[col];
#pragma unroll
        for (int i = 0; i < 8; ++i) {
            int row = row0 + i + rowadd;
            outp[(size_t)row * GC + col] = acc0[ct][i] + bb;
            if (has2)
                outp[(size_t)(row + 16) * GC + col] = acc1[ct][i] + bb;
        }
    }
}

// one edge per wave32: 128-wide dot(q[dst], k[src]) via float4 + butterfly reduce
__global__ void alpha_max_kernel(const float* __restrict__ q, const float* __restrict__ k,
                                 const int* __restrict__ src, const int* __restrict__ dst,
                                 float* __restrict__ alpha, unsigned* __restrict__ amax) {
    int e = blockIdx.x * 8 + (threadIdx.x >> 5);
    int lane = threadIdx.x & 31;
    int s = src[e], d = dst[e];
    const float4* qp = (const float4*)(q + (size_t)d * GC);
    const float4* kp = (const float4*)(k + (size_t)s * GC);
    float4 a4 = qp[lane], b4 = kp[lane];
    float p = a4.x * b4.x + a4.y * b4.y + a4.z * b4.z + a4.w * b4.w;
#pragma unroll
    for (int o = 16; o > 0; o >>= 1) p += __shfl_xor(p, o, 32);
    if (lane == 0) {
        p *= 0.0883883476483184f;   // 1/sqrt(128)
        alpha[e] = p;
        atomicMax(amax + d, enc_ord(p));
    }
}

__global__ void exp_denom_kernel(float* __restrict__ alpha, const unsigned* __restrict__ amax,
                                 const int* __restrict__ dst, float* __restrict__ denom) {
    int e = blockIdx.x * blockDim.x + threadIdx.x;
    if (e >= GE) return;
    int d = dst[e];
    float w = expf(alpha[e] - dec_ord(amax[d]));
    alpha[e] = w;
    atomicAdd(denom + d, w);
}

// one edge per wave32: h[dst] += (w/denom[dst]) * v[src]  (h pre-loaded with skip)
__global__ void message_kernel(const float* __restrict__ alpha, const float* __restrict__ denom,
                               const float* __restrict__ v, const int* __restrict__ src,
                               const int* __restrict__ dst, float* __restrict__ h) {
    int e = blockIdx.x * 8 + (threadIdx.x >> 5);
    int lane = threadIdx.x & 31;
    int s = src[e], d = dst[e];
    float wgt = alpha[e] / denom[d];
    const float4* vp = (const float4*)(v + (size_t)s * GC);
    float4 m4 = vp[lane];
    float* hp = h + (size_t)d * GC + lane * 4;
    atomicAdd(hp + 0, wgt * m4.x);
    atomicAdd(hp + 1, wgt * m4.y);
    atomicAdd(hp + 2, wgt * m4.z);
    atomicAdd(hp + 3, wgt * m4.w);
}

// re-quantize h -> bf16 for next layer; accumulate per-graph mean pool into out
__global__ void finalize_kernel(const float* __restrict__ h, unsigned short* __restrict__ hbf,
                                const int* __restrict__ bid, const float* __restrict__ counts,
                                float* __restrict__ out, int layer) {
    int t = blockIdx.x * blockDim.x + threadIdx.x;
    if (t >= GN * GC) return;
    int n = t >> 7, c = t & 127;
    float val = h[t];
    hbf[t] = f2bf(val);
    int b = bid[n];
    float cnt = fmaxf(counts[b], 1.0f);
    atomicAdd(out + (size_t)b * (GDEPTH * GC) + layer * GC + c, val / cnt);
}

extern "C" void kernel_launch(void* const* d_in, const int* in_sizes, int n_in,
                              void* d_out, int out_size, void* d_ws, size_t ws_size,
                              hipStream_t stream) {
    const float* x    = (const float*)d_in[0];
    const int*   ei   = (const int*)d_in[1];
    const int*   src  = ei;
    const int*   dst  = ei + GE;
    const int*   bid  = (const int*)d_in[2];
    const float* encW = (const float*)d_in[3];
    const float* encb = (const float*)d_in[4];
    const float* Wq   = (const float*)d_in[5];
    const float* bq   = (const float*)d_in[6];
    const float* Wk   = (const float*)d_in[7];
    const float* bk   = (const float*)d_in[8];
    const float* Wv   = (const float*)d_in[9];
    const float* bv   = (const float*)d_in[10];
    const float* Ws   = (const float*)d_in[11];
    const float* bs   = (const float*)d_in[12];
    float* out = (float*)d_out;

    // carve workspace (256B aligned slices)
    size_t off = 0;
    char* base = (char*)d_ws;
    auto carve = [&](size_t bytes) -> void* {
        void* p = base + off;
        off = (off + bytes + 255) & ~(size_t)255;
        return p;
    };
    float*          h      = (float*)carve((size_t)GN * GC * 4);
    unsigned short* hbf    = (unsigned short*)carve((size_t)GN * GC * 2);
    float*          q      = (float*)carve((size_t)GN * GC * 4);
    float*          kbuf   = (float*)carve((size_t)GN * GC * 4);
    float*          v      = (float*)carve((size_t)GN * GC * 4);
    float*          alpha  = (float*)carve((size_t)GE * 4);
    unsigned*       amax   = (unsigned*)carve((size_t)GN * 4);
    float*          denom  = (float*)carve((size_t)GN * 4);
    float*          counts = (float*)carve((size_t)GB * 4);
    unsigned short* wpack  = (unsigned short*)carve((size_t)GDEPTH * 4 * GC * GC * 2);

    // init: zero output + counts, per-graph node counts, pack weights to bf16 frags
    zero_f32_kernel<<<(GB * GDEPTH * GC + 255) / 256, 256, 0, stream>>>(out, GB * GDEPTH * GC);
    zero_f32_kernel<<<1, 256, 0, stream>>>(counts, GB);
    counts_kernel<<<(GN + 255) / 256, 256, 0, stream>>>(bid, counts);
    pack_weights_kernel<<<(GDEPTH * 4 * GC * GC + 255) / 256, 256, 0, stream>>>(Wq, Wk, Wv, Ws, wpack);

    // feature encoder: h = concat(x, x@encW + encb), plus bf16 copy
    encoder_kernel<<<(GN * GC + 255) / 256, 256, 0, stream>>>(x, encW, encb, h, hbf);

    for (int l = 0; l < GDEPTH; ++l) {
        init_layer_kernel<<<(GN + 255) / 256, 256, 0, stream>>>(denom, amax);
        // q,k,v from hbf; skip (x@Ws+bs) written straight into h
        gemm_qkvs_kernel<<<(GN + 31) / 32, 128, 0, stream>>>(hbf, wpack, l, q, kbuf, v, h,
                                                             bq + l * GC, bk + l * GC,
                                                             bv + l * GC, bs + l * GC);
        alpha_max_kernel<<<GE / 8, 256, 0, stream>>>(q, kbuf, src, dst, alpha, amax);
        exp_denom_kernel<<<(GE + 255) / 256, 256, 0, stream>>>(alpha, amax, dst, denom);
        message_kernel<<<GE / 8, 256, 0, stream>>>(alpha, denom, v, src, dst, h);
        finalize_kernel<<<(GN * GC + 255) / 256, 256, 0, stream>>>(h, hbf, bid, counts, out, l);
    }
}
